// HardRuleList_82557861363924
// MI455X (gfx1250) — matrix operand order repaired and checked by reading, not compiled, not run on hardware
//
#include <hip/hip_runtime.h>

#define NS 32768
#define NF 64
#define NR 256
#define NC 10
#define WSTRIDE 264   // padded half-stride between class columns (528 B -> bank-spread)

typedef __attribute__((ext_vector_type(16))) _Float16 v16h;
typedef __attribute__((ext_vector_type(8)))  _Float16 v8h;
typedef __attribute__((ext_vector_type(8)))  float    v8f;
typedef int v4i_b __attribute__((vector_size(16)));   // matches builtin pointee type

#define AS1 __attribute__((address_space(1)))
#define AS3 __attribute__((address_space(3)))

#if defined(__has_builtin)
#if __has_builtin(__builtin_amdgcn_global_load_async_to_lds_b128)
#define USE_ASYNC_LDS 1
#else
#define USE_ASYNC_LDS 0
#endif
#else
#define USE_ASYNC_LDS 0
#endif

// ---------------------------------------------------------------------------
// Prep: stable descending rank of each rule + branch-free ordered rule table:
//   inactive feature (and_weight <= 0)   -> bounds (-inf, +inf)  (always sat)
//   invalid rule (all features inactive) -> bounds (+inf, -inf)  (never fires)
// ord_rid[pos] = original rule id at priority position pos.
// ---------------------------------------------------------------------------
__global__ void __launch_bounds__(256)
rule_prep_kernel(const float* __restrict__ cut,   // (F,2,R)
                 const float* __restrict__ aw,    // (R,F)
                 const float* __restrict__ ro,    // (R)
                 float* __restrict__ ord_lo,      // (R,F)
                 float* __restrict__ ord_hi,      // (R,F)
                 int*   __restrict__ ord_rid)     // (R)
{
    const int r = threadIdx.x;
    if (r >= NR) return;

    const float my = ro[r];
    int pos = 0;
    for (int j = 0; j < NR; ++j) {
        const float oj = ro[j];
        pos += (oj > my) || ((oj == my) && (j < r));   // stable descending rank
    }

    bool any_active = false;
    for (int f = 0; f < NF; ++f)
        any_active = any_active || (aw[r * NF + f] > 0.0f);

    for (int f = 0; f < NF; ++f) {
        float lo = cut[f * 2 * NR + r];            // cut_points[f][0][r]
        float hi = cut[f * 2 * NR + NR + r];       // cut_points[f][1][r]
        if (!(aw[r * NF + f] > 0.0f)) {            // inactive feature
            lo = -INFINITY;
            hi =  INFINITY;
        }
        ord_lo[pos * NF + f] = lo;
        ord_hi[pos * NF + f] = hi;
    }
    if (!any_active) {                              // invalid rule never fires
        ord_lo[pos * NF] =  INFINITY;
        ord_hi[pos * NF] = -INFINITY;
    }
    ord_rid[pos] = r;
}

// ---------------------------------------------------------------------------
// Main: one sample per thread, 256 threads (8 wave32) per block.
// Rule table DMA'd into LDS (async-to-LDS when available, overlapped with
// weight-split math and x loads); priority scan with wave early-exit; final
// prediction via one-hot x rule_weights with v_wmma_f32_16x16x32_f16 using an
// f16 hi/lo split of the fp32 weights (near-exact fp32 result).
// ---------------------------------------------------------------------------
__global__ void __launch_bounds__(256)
rulelist_main_kernel(const float* __restrict__ x,        // (N,F)
                     const float* __restrict__ rw,       // (R,C)
                     const float* __restrict__ ord_lo,   // (R,F) ord_hi contiguous after
                     const float* __restrict__ ord_hi,   // (R,F)
                     const int*   __restrict__ ord_rid,  // (R)
                     float* __restrict__ out)            // (N,C)
{
    extern __shared__ float smem[];
    float*    s_lo  = smem;                                  // R*F floats
    float*    s_hi  = smem + NR * NF;                        // R*F floats (contiguous)
    _Float16* s_whi = (_Float16*)(smem + 2 * NR * NF);       // 16*WSTRIDE halves (transposed)
    _Float16* s_wlo = s_whi + 16 * WSTRIDE;                  // 16*WSTRIDE halves
    int*      s_rid = (int*)(s_wlo + 16 * WSTRIDE);          // 256 ints

    const int tid = threadIdx.x;

#if USE_ASYNC_LDS
    // Fire-and-forget DMA of the 128 KB bounds table (ord_lo||ord_hi) into
    // LDS: 32 x 16B per thread, tracked on ASYNCcnt. Overlaps with the weight
    // split and x register loads below; waited just before the barrier.
    {
        const uint32_t  lbase = (uint32_t)(uintptr_t)(void*)s_lo;   // flat->LDS offset (addr[31:0])
        const uintptr_t gbase = (uintptr_t)(const void*)ord_lo;
        #pragma unroll 4
        for (int i = 0; i < 32; ++i) {
            const uint32_t off = ((uint32_t)i * 256u + (uint32_t)tid) * 16u;
            AS1 v4i_b* g = (AS1 v4i_b*)(gbase + (uintptr_t)off);
            AS3 v4i_b* l = (AS3 v4i_b*)(uint32_t)(lbase + off);
            __builtin_amdgcn_global_load_async_to_lds_b128(g, l, 0, 0);
        }
    }
#endif

    // Stage padded class weights, TRANSPOSED (s_w[class][rule], stride 264
    // halves) as an f16 hi/lo pair: w == hi + lo to ~2^-22 relative.
    {
        #pragma unroll
        for (int c = 0; c < 16; ++c) {
            const float w  = (c < NC) ? rw[tid * NC + c] : 0.0f;
            const _Float16 whi = (_Float16)w;
            const _Float16 wlo = (_Float16)(w - (float)whi);
            s_whi[c * WSTRIDE + tid] = whi;
            s_wlo[c * WSTRIDE + tid] = wlo;
        }
    }

#if !USE_ASYNC_LDS
    // Fallback: synchronous staging of ordered rule bounds via float4.
    {
        const float4* glo = (const float4*)ord_lo;
        const float4* ghi = (const float4*)ord_hi;
        float4* slo4 = (float4*)s_lo;
        float4* shi4 = (float4*)s_hi;
        for (int i = tid; i < (NR * NF) / 4; i += 256) {
            slo4[i] = glo[i];
            shi4[i] = ghi[i];
        }
    }
#endif

    // Load this thread's sample into registers (64 floats as 16x float4).
    const int s = blockIdx.x * 256 + tid;
    float4 xv[16];
    const float4* gx = (const float4*)(x + (size_t)s * NF);
    #pragma unroll
    for (int i = 0; i < 16; ++i) xv[i] = gx[i];

#if USE_ASYNC_LDS
#if __has_builtin(__builtin_amdgcn_s_wait_asynccnt)
    __builtin_amdgcn_s_wait_asynccnt(0);
#else
    asm volatile("s_wait_asynccnt 0" ::: "memory");
#endif
#endif
    __syncthreads();

    // Priority-ordered scan: first firing rule position.
    int found = -1;
    for (int p = 0; p < NR; ++p) {
        const float4* lo4 = (const float4*)(s_lo + p * NF);
        const float4* hi4 = (const float4*)(s_hi + p * NF);
        bool fire = true;
        #pragma unroll
        for (int i = 0; i < 16; ++i) {
            const float4 lo = lo4[i];
            const float4 hi = hi4[i];
            fire = fire & (xv[i].x > lo.x) & (xv[i].x < hi.x)
                        & (xv[i].y > lo.y) & (xv[i].y < hi.y)
                        & (xv[i].z > lo.z) & (xv[i].z < hi.z)
                        & (xv[i].w > lo.w) & (xv[i].w < hi.w);
        }
        if (fire && found < 0) found = p;
        if (((p & 7) == 7) && __all(found >= 0)) break;   // wave early-exit
    }

    // Map priority position back to original rule id (-1 => no rule fired).
    s_rid[tid] = (found >= 0) ? ord_rid[found] : -1;
    __syncthreads();

    // ---- WMMA one-hot gather: D(16x16) = A(16x256 one-hot) * B(256x16) ----
    const int  lane   = tid & 31;
    const int  wv     = tid >> 5;
    const int  n      = lane & 15;          // class column (and A row index)
    const bool hiHalf = (lane >= 16);
    const _Float16 h1 = (_Float16)1.0f;
    const _Float16 h0 = (_Float16)0.0f;

    #pragma unroll
    for (int t = 0; t < 2; ++t) {           // two 16-sample tiles per wave
        const int row0 = wv * 32 + t * 16;
        const int r_m  = s_rid[row0 + n];   // rule id for A-matrix row (lane&15)
        v8f c = {};

        for (int k0 = 0; k0 < NR; k0 += 32) {
            const int  rel = r_m - k0;
            const bool in  = (rel >= 0) && (rel < 32);
            if (!__any(in)) continue;       // wave-uniform chunk skip

            // A: 16x32 f16 one-hot. Lane layout (contiguous k order chosen here, same as B):
            //   element h -> K = h + 16*[lane>=16]
            v16h a;
            #pragma unroll
            for (int h = 0; h < 16; ++h) {
                const int K = h + (hiHalf ? 16 : 0);
                a[h] = (rel == K) ? h1 : h0;
            }
            // B: 32x16 f16. element h -> K = h + 16*[lane>=16], N = lane&15.
            // Transposed LDS layout makes each lane's 16 halves contiguous.
            const int koff = n * WSTRIDE + k0 + (hiHalf ? 16 : 0);
            const v8h* ph = (const v8h*)(s_whi + koff);
            const v8h* pl = (const v8h*)(s_wlo + koff);
            const v16h bh = __builtin_shufflevector(ph[0], ph[1],
                0, 1, 2, 3, 4, 5, 6, 7, 8, 9, 10, 11, 12, 13, 14, 15);
            const v16h bl = __builtin_shufflevector(pl[0], pl[1],
                0, 1, 2, 3, 4, 5, 6, 7, 8, 9, 10, 11, 12, 13, 14, 15);

            c = __builtin_amdgcn_wmma_f32_16x16x32_f16(false, a, false, bh,
                                                       (short)0, c, false, false);
            c = __builtin_amdgcn_wmma_f32_16x16x32_f16(false, a, false, bl,
                                                       (short)0, c, false, false);
        }

        // D layout: VGPR j holds row M = j + 8*[lane>=16], col N = lane&15.
        const int sbase = blockIdx.x * 256 + row0;
        if (n < NC) {
            #pragma unroll
            for (int j = 0; j < 8; ++j) {
                const int m = j + (hiHalf ? 8 : 0);
                out[(size_t)(sbase + m) * NC + n] = c[j];
            }
        }
    }
}

// ---------------------------------------------------------------------------
extern "C" void kernel_launch(void* const* d_in, const int* in_sizes, int n_in,
                              void* d_out, int out_size, void* d_ws, size_t ws_size,
                              hipStream_t stream) {
    const float* x   = (const float*)d_in[0];   // (N,F)
    const float* cut = (const float*)d_in[1];   // (F,2,R)
    const float* aw  = (const float*)d_in[2];   // (R,F)
    const float* ro  = (const float*)d_in[3];   // (R)
    const float* rw  = (const float*)d_in[4];   // (R,C)
    float* out = (float*)d_out;

    float* ord_lo = (float*)d_ws;               // R*F floats
    float* ord_hi = ord_lo + NR * NF;           // R*F floats (contiguous after ord_lo)
    int*   ord_rid = (int*)(ord_hi + NR * NF);  // R ints

    rule_prep_kernel<<<1, 256, 0, stream>>>(cut, aw, ro, ord_lo, ord_hi, ord_rid);

    // LDS: 2*R*F*4 (bounds) + 2*16*264*2 (f16 weights) + 256*4 (rids)
    const size_t smem_bytes = (size_t)2 * NR * NF * 4 + (size_t)2 * 16 * WSTRIDE * 2 + 256 * 4;
    rulelist_main_kernel<<<NS / 256, 256, smem_bytes, stream>>>(
        x, rw, ord_lo, ord_hi, ord_rid, out);
}
